// Attention_66915590471696
// MI455X (gfx1250) — matrix operand order repaired and checked by reading, not compiled, not run on hardware
//
#include <hip/hip_runtime.h>

typedef __bf16 bf16;
typedef __attribute__((ext_vector_type(16))) __bf16 bf16x16;
typedef __attribute__((ext_vector_type(8)))  __bf16 bf16x8;
typedef __attribute__((ext_vector_type(8)))  float  f32x8;

#define Bb 2
#define Nn 2048
#define Dd 1024
#define Hh 16
#define DHd 64
#define MROWS (Bb*Nn)          // 4096
static __device__ __constant__ float kScale = 0.125f; // 64^-0.5

// ---------------- prep: fp32 -> bf16 copy ----------------
__global__ void k_cvt_bf16(const float* __restrict__ src, bf16* __restrict__ dst, int n) {
    int i = blockIdx.x * blockDim.x + threadIdx.x;
    if (i < n) dst[i] = (bf16)src[i];
}

// W[K x Nc] row-major -> Wt[Nc x K] bf16 (so WMMA B-fragments are contiguous)
__global__ void k_transpose_bf16(const float* __restrict__ W, bf16* __restrict__ Wt,
                                 int K, int Nc) {
    int i = blockIdx.x * blockDim.x + threadIdx.x;
    if (i >= K * Nc) return;
    int k = i / Nc, n = i % Nc;
    Wt[(size_t)n * K + k] = (bf16)W[i];
}

// ---------------- mix = sigmoid(x @ Wmix) ----------------
__global__ void k_mix(const float* __restrict__ x, const float* __restrict__ Wmix,
                      float* __restrict__ mix) {
    int i = blockIdx.x * blockDim.x + threadIdx.x;   // over (B*N)*H
    if (i >= MROWS * Hh) return;
    int h = i % Hh;
    int row = i / Hh;
    const float* xr = x + (size_t)row * Dd;
    float acc = 0.f;
    for (int k = 0; k < Dd; ++k) acc += xr[k] * Wmix[k * Hh + h];
    mix[i] = 1.f / (1.f + expf(-acc));
}

// ---------------- generic bf16 WMMA GEMM: C = A @ Bt^T (+bias) ----------------
// A:  [M x K] bf16 row-major
// Bt: [Nc x K] bf16 (B transposed) so per-lane B-fragment loads are contiguous
// one wave per 64x64 tile, K-step 32, 16 v_wmma per step
__global__ __launch_bounds__(32)
void k_gemm_bf16(const bf16* __restrict__ A, const bf16* __restrict__ Bt,
                 const float* __restrict__ bias, float* __restrict__ C,
                 int M, int Nc, int K) {
    const int lane = threadIdx.x & 31;
    const int half = lane >> 4;       // 0/1
    const int l16  = lane & 15;
    const int tn = blockIdx.x * 64;
    const int tm = blockIdx.y * 64;
    (void)M;

    f32x8 acc[4][4];
    for (int mi = 0; mi < 4; ++mi)
        for (int ni = 0; ni < 4; ++ni)
            acc[mi][ni] = {};

    const int koff = half * 16;       // A-fragment K sub-offset (contiguous order k = 16*half + s)
    for (int kc = 0; kc < K; kc += 32) {
        // Speculative prefetch of the next K tile (translation failures are
        // silently dropped on CDNA5, so no end-of-buffer guard: branch-free loop).
        __builtin_prefetch(A  + (size_t)(tm + l16) * K + kc + 32, 0, 1);
        __builtin_prefetch(Bt + (size_t)(tn + l16) * K + kc + 32, 0, 1);

        bf16x16 a[4], b[4];
        for (int mi = 0; mi < 4; ++mi) {
            const bf16* ar = A + (size_t)(tm + mi * 16 + l16) * K + kc + koff;
            bf16x8 lo = *(const bf16x8*)(ar);        // K = koff..koff+7
            bf16x8 hi = *(const bf16x8*)(ar + 8);    // K = koff+8..koff+15
            a[mi] = __builtin_shufflevector(lo, hi,
                     0,1,2,3,4,5,6,7,8,9,10,11,12,13,14,15);
        }
        for (int ni = 0; ni < 4; ++ni) {
            const bf16* br = Bt + (size_t)(tn + ni * 16 + l16) * K + kc + half * 16;
            b[ni] = *(const bf16x16*)(br);           // 16 contiguous K values
        }
        for (int mi = 0; mi < 4; ++mi)
            for (int ni = 0; ni < 4; ++ni)
                acc[mi][ni] = __builtin_amdgcn_wmma_f32_16x16x32_bf16(
                    false, a[mi], false, b[ni], (short)0, acc[mi][ni], false, false);
    }

    for (int mi = 0; mi < 4; ++mi)
        for (int ni = 0; ni < 4; ++ni)
            for (int r = 0; r < 8; ++r) {
                int row = tm + mi * 16 + r + half * 8;   // C layout: VGPR r, half picks M / M+8
                int col = tn + ni * 16 + l16;
                float v = acc[mi][ni][r];
                if (bias) v += bias[col];
                C[(size_t)row * Nc + col] = v;
            }
}

// ---------------- RoPE on q/k + value-residual lerp; emit bf16 Q,K,Vt ----------------
__global__ void k_rope_v(const float* __restrict__ Cq, const float* __restrict__ Ckv,
                         const float* __restrict__ rot, const float* __restrict__ mix,
                         const float* __restrict__ vres,
                         bf16* __restrict__ Qr, bf16* __restrict__ Kr, bf16* __restrict__ Vt) {
    int i = blockIdx.x * blockDim.x + threadIdx.x;   // B*N*H*DH
    if (i >= Bb * Nn * Hh * DHd) return;
    int d = i & 63;
    int h = (i >> 6) & 15;
    int n = (i >> 10) & 2047;
    int b = i >> 21;

    float ang = rot[n * DHd + d];
    float c = cosf(ang), s = sinf(ang);

    size_t rowq = (size_t)b * Nn + n;
    const float* q  = Cq  + rowq * (Hh * DHd) + h * DHd;
    const float* kv = Ckv + rowq * (2 * Hh * DHd);
    const float* kk = kv + h * DHd;

    float qrot = (d < 32) ? -q[d + 32]  : q[d - 32];
    float krot = (d < 32) ? -kk[d + 32] : kk[d - 32];
    float qn = q[d]  * c + qrot * s;
    float kn = kk[d] * c + krot * s;

    size_t bhn = ((size_t)(b * Hh + h) * Nn + n);
    Qr[bhn * DHd + d] = (bf16)qn;
    Kr[bhn * DHd + d] = (bf16)kn;

    float vv = kv[Hh * DHd + h * DHd + d];
    float mx = mix[((size_t)b * Nn + n) * Hh + h];
    float rs = vres[bhn * DHd + d];
    vv = vv + mx * (rs - vv);
    Vt[((size_t)(b * Hh + h) * DHd + d) * Nn + n] = (bf16)vv;   // transposed [b,h,d,n]
}

// ---------------- flash attention: one wave per (b,h,16-query tile) ----------------
__global__ __launch_bounds__(32)
void k_flash(const bf16* __restrict__ Qr, const bf16* __restrict__ Kr,
             const bf16* __restrict__ Vt, const float* __restrict__ bias,
             const unsigned char* __restrict__ mask, bf16* __restrict__ AO) {
    __shared__ __align__(32) bf16 lds_p[16 * 32];   // per-block P tile (bf16)

    const int lane = threadIdx.x & 31;
    const int half = lane >> 4;
    const int l16  = lane & 15;
    const int qt = blockIdx.x * 16;
    const int h  = blockIdx.y;
    const int b  = blockIdx.z;
    const size_t bh = (size_t)(b * Hh + h);

    // Q A-fragments (16x64 split into two 16x32)
    bf16x16 qa[2];
    {
        const bf16* qbase = Qr + (bh * Nn + qt + l16) * DHd + half * 16;
        for (int f = 0; f < 2; ++f) {
            bf16x8 lo = *(const bf16x8*)(qbase + f * 32);
            bf16x8 hi = *(const bf16x8*)(qbase + f * 32 + 8);
            qa[f] = __builtin_shufflevector(lo, hi,
                     0,1,2,3,4,5,6,7,8,9,10,11,12,13,14,15);
        }
    }

    f32x8 o[4];
    for (int ni = 0; ni < 4; ++ni) o[ni] = {};
    float m[8], l[8];
    for (int r = 0; r < 8; ++r) { m[r] = -3.0e38f; l[r] = 0.f; }

    const float* biash = bias + (size_t)h * Nn * Nn;

    for (int jc = 0; jc < Nn; jc += 32) {
        // Speculative prefetch of next chunk's K rows, V rows, bias row segment.
        __builtin_prefetch(Kr + (bh * Nn + jc + 32 + l16) * DHd, 0, 1);
        __builtin_prefetch(Vt + (bh * DHd + l16) * Nn + jc + 32, 0, 1);
        __builtin_prefetch(biash + (size_t)(qt + l16) * Nn + jc + 32, 0, 1);

        // ---- sim = Q @ K^T (two 16x16 tiles) ----
        f32x8 sim[2]; sim[0] = {}; sim[1] = {};
        for (int sub = 0; sub < 2; ++sub) {
            int key = jc + sub * 16 + l16;
            const bf16* kb = Kr + (bh * Nn + key) * DHd + half * 16;
            bf16x16 b0 = *(const bf16x16*)(kb);        // d = 0..31 slice
            bf16x16 b1 = *(const bf16x16*)(kb + 32);   // d = 32..63 slice
            sim[sub] = __builtin_amdgcn_wmma_f32_16x16x32_bf16(
                false, qa[0], false, b0, (short)0, sim[sub], false, false);
            sim[sub] = __builtin_amdgcn_wmma_f32_16x16x32_bf16(
                false, qa[1], false, b1, (short)0, sim[sub], false, false);
        }

        // ---- scale + bias + mask, online softmax ----
        float p[2][8], cmax[8];
        for (int r = 0; r < 8; ++r) cmax[r] = -3.0e38f;
        for (int sub = 0; sub < 2; ++sub) {
            int key = jc + sub * 16 + l16;
            bool mk = mask[(size_t)b * Nn + key] != 0;
            for (int r = 0; r < 8; ++r) {
                int qrow = qt + r + half * 8;
                float v = sim[sub][r] * kScale + biash[(size_t)qrow * Nn + key];
                if (!mk) v = -3.0e38f;
                p[sub][r] = v;
                cmax[r] = fmaxf(cmax[r], v);
            }
        }
        for (int r = 0; r < 8; ++r)
            for (int off = 8; off >= 1; off >>= 1)
                cmax[r] = fmaxf(cmax[r], __shfl_xor(cmax[r], off, 32));

        float corr[8];
        for (int r = 0; r < 8; ++r) {
            float mn = fmaxf(m[r], cmax[r]);
            corr[r] = expf(m[r] - mn);
            m[r] = mn;
        }
        float rsum[8];
        for (int r = 0; r < 8; ++r) rsum[r] = 0.f;
        for (int sub = 0; sub < 2; ++sub)
            for (int r = 0; r < 8; ++r) {
                float e = expf(p[sub][r] - m[r]);
                p[sub][r] = e;
                rsum[r] += e;
            }
        for (int r = 0; r < 8; ++r) {
            for (int off = 8; off >= 1; off >>= 1)
                rsum[r] += __shfl_xor(rsum[r], off, 32);
            l[r] = l[r] * corr[r] + rsum[r];
        }
        for (int ni = 0; ni < 4; ++ni)
            for (int r = 0; r < 8; ++r)
                o[ni][r] *= corr[r];

        // ---- C-layout -> A-layout via LDS (bf16) ----
        __syncthreads();
        for (int sub = 0; sub < 2; ++sub)
            for (int r = 0; r < 8; ++r)
                lds_p[(r + half * 8) * 32 + sub * 16 + l16] = (bf16)p[sub][r];
        __syncthreads();

        bf16x16 pa;
        {
            const bf16* pr = lds_p + l16 * 32 + half * 16;
            bf16x8 lo = *(const bf16x8*)(pr);
            bf16x8 hi = *(const bf16x8*)(pr + 8);
            pa = __builtin_shufflevector(lo, hi,
                  0,1,2,3,4,5,6,7,8,9,10,11,12,13,14,15);
        }

        // ---- O += P @ V (V transposed: keys contiguous per lane) ----
        for (int ni = 0; ni < 4; ++ni) {
            const bf16* vb = Vt + (bh * DHd + ni * 16 + l16) * Nn + jc + half * 16;
            bf16x16 bv = *(const bf16x16*)(vb);
            o[ni] = __builtin_amdgcn_wmma_f32_16x16x32_bf16(
                false, pa, false, bv, (short)0, o[ni], false, false);
        }
    }

    // ---- normalize, emit bf16 for the output projection ----
    for (int ni = 0; ni < 4; ++ni)
        for (int r = 0; r < 8; ++r) {
            int row = qt + r + half * 8;
            int col = h * DHd + ni * 16 + l16;
            AO[((size_t)b * Nn + row) * (Hh * DHd) + col] = (bf16)(o[ni][r] / l[r]);
        }
}

// ---------------- host: orchestrate ----------------
extern "C" void kernel_launch(void* const* d_in, const int* in_sizes, int n_in,
                              void* d_out, int out_size, void* d_ws, size_t ws_size,
                              hipStream_t stream) {
    (void)in_sizes; (void)n_in; (void)out_size; (void)ws_size;
    const float*         x    = (const float*)d_in[0];
    const unsigned char* mask = (const unsigned char*)d_in[1]; // jnp bool -> 1 byte
    const float*         rot  = (const float*)d_in[2];
    const float*         abia = (const float*)d_in[3];
    const float*         vres = (const float*)d_in[4];
    const float*         Wq   = (const float*)d_in[5];
    const float*         Wkv  = (const float*)d_in[6];
    const float*         Wmix = (const float*)d_in[7];
    const float*         Wout = (const float*)d_in[8];
    const float*         bout = (const float*)d_in[9];
    float*               out  = (float*)d_out;

    char* wsp = (char*)d_ws;
    auto carve = [&](size_t bytes) {
        char* p = wsp;
        wsp += (bytes + 255) & ~(size_t)255;
        return p;
    };
    bf16*  xb    = (bf16*)carve((size_t)MROWS * Dd * 2);
    bf16*  wqT   = (bf16*)carve((size_t)Dd * Dd * 2);
    bf16*  wkvT  = (bf16*)carve((size_t)2 * Dd * Dd * 2);
    bf16*  woutT = (bf16*)carve((size_t)Dd * Dd * 2);
    float* cq    = (float*)carve((size_t)MROWS * Dd * 4);
    float* ckv   = (float*)carve((size_t)MROWS * 2 * Dd * 4);
    float* mix   = (float*)carve((size_t)MROWS * Hh * 4);
    bf16*  qr    = (bf16*)carve((size_t)MROWS * Dd * 2);
    bf16*  kr    = (bf16*)carve((size_t)MROWS * Dd * 2);
    bf16*  vt    = (bf16*)carve((size_t)MROWS * Dd * 2);
    bf16*  ao    = (bf16*)carve((size_t)MROWS * Dd * 2);

    const int T = 256;
    // 1) conversions
    {
        int n = MROWS * Dd;
        k_cvt_bf16<<<(n + T - 1) / T, T, 0, stream>>>(x, xb, n);
        n = Dd * Dd;
        k_transpose_bf16<<<(n + T - 1) / T, T, 0, stream>>>(Wq, wqT, Dd, Dd);
        n = Dd * 2 * Dd;
        k_transpose_bf16<<<(n + T - 1) / T, T, 0, stream>>>(Wkv, wkvT, Dd, 2 * Dd);
        n = Dd * Dd;
        k_transpose_bf16<<<(n + T - 1) / T, T, 0, stream>>>(Wout, woutT, Dd, Dd);
    }
    // 2) mix gate
    {
        int n = MROWS * Hh;
        k_mix<<<(n + T - 1) / T, T, 0, stream>>>(x, Wmix, mix);
    }
    // 3) projections (WMMA GEMMs)
    k_gemm_bf16<<<dim3(Dd / 64, MROWS / 64), 32, 0, stream>>>(
        xb, wqT, nullptr, cq, MROWS, Dd, Dd);
    k_gemm_bf16<<<dim3(2 * Dd / 64, MROWS / 64), 32, 0, stream>>>(
        xb, wkvT, nullptr, ckv, MROWS, 2 * Dd, Dd);
    // 4) RoPE + value lerp, emit bf16 Q/K/Vt
    {
        int n = Bb * Nn * Hh * DHd;
        k_rope_v<<<(n + T - 1) / T, T, 0, stream>>>(cq, ckv, rot, mix, vres, qr, kr, vt);
    }
    // 5) flash attention
    k_flash<<<dim3(Nn / 16, Hh, Bb), 32, 0, stream>>>(qr, kr, vt, abia, mask, ao);
    // 6) output projection + bias
    k_gemm_bf16<<<dim3(Dd / 64, MROWS / 64), 32, 0, stream>>>(
        ao, woutT, bout, out, MROWS, Dd, Dd);
}
